// GatedAttention_360777253298
// MI455X (gfx1250) — matrix operand order repaired and checked by reading, not compiled, not run on hardware
//
#include <hip/hip_runtime.h>
#include <hip/hip_bf16.h>
#include <math.h>

typedef __attribute__((ext_vector_type(16))) _Float16 v16h;
typedef __attribute__((ext_vector_type(8)))  float    v8f;

union FragH { v16h v; float4 f4[2]; };

#define WMMA_F32_F16(a, b, c) \
    __builtin_amdgcn_wmma_f32_16x16x32_f16(false, (a), false, (b), (short)0, (c), false, false)

constexpr int S_LEN = 2048;
constexpr int NB    = 2;
constexpr int E_DIM = 2048;
constexpr int NH    = 16;
constexpr int NKV   = 2;
constexpr int DH    = 256;
constexpr int GQ    = NH / NKV;       // 8
constexpr int MROWS = NB * S_LEN;     // 4096

// ---------------------------------------------------------------------------
// CDNA5 async global->LDS copy (ASYNCcnt path), with sync fallback.
// ---------------------------------------------------------------------------
#if defined(__HIP_DEVICE_COMPILE__) && __has_builtin(__builtin_amdgcn_global_load_async_to_lds_b128)
#define HAS_ASYNC_LDS 1
typedef int v4si __attribute__((vector_size(16)));
typedef __attribute__((address_space(1))) v4si gv4si;
typedef __attribute__((address_space(3))) v4si lv4si;
#else
#define HAS_ASYNC_LDS 0
#endif

__device__ __forceinline__ void copy16_to_lds(const _Float16* g, _Float16* l)
{
#if HAS_ASYNC_LDS
    __builtin_amdgcn_global_load_async_to_lds_b128(
        (gv4si*)(uintptr_t)g, (lv4si*)(uint32_t)(uintptr_t)l, 0, 0);
#else
    *(float4*)l = *(const float4*)g;
#endif
}

__device__ __forceinline__ void async_wait_all()
{
#if HAS_ASYNC_LDS
    asm volatile("s_wait_asynccnt 0x0" ::: "memory");
#endif
}

// ---------------------------------------------------------------------------
// Prepass kernels: f32->f16 convert, f32->f16 transpose (weights), V transpose.
// ---------------------------------------------------------------------------
__global__ void convert_f16_kernel(const float* __restrict__ src,
                                   _Float16* __restrict__ dst, long n)
{
    long i = (long)blockIdx.x * blockDim.x + threadIdx.x;
    if (i < n) dst[i] = (_Float16)src[i];
}

// Wt[n][k] = (f16) W[k][n];  W is K x N row-major.
__global__ void transpose_f16_kernel(const float* __restrict__ W,
                                     _Float16* __restrict__ Wt, int K, int N)
{
    long i = (long)blockIdx.x * blockDim.x + threadIdx.x;
    if (i >= (long)K * N) return;
    int kk = (int)(i / N);
    int nn = (int)(i % N);
    Wt[(size_t)nn * K + kk] = (_Float16)W[i];
}

// vt[(b*NKV+kvh)*DH + d][s] = v[(b*S+s)][kvh*DH + d]   (f16 -> f16, d-major)
__global__ void vtranspose_kernel(const _Float16* __restrict__ v,
                                  _Float16* __restrict__ vt)
{
    long i = (long)blockIdx.x * blockDim.x + threadIdx.x;
    long total = (long)NB * NKV * DH * S_LEN;
    if (i >= total) return;
    int s   = (int)(i % S_LEN);
    int d   = (int)((i / S_LEN) % DH);
    int bk  = (int)(i / ((long)S_LEN * DH));
    int kvh = bk % NKV;
    int b   = bk / NKV;
    vt[i] = v[(size_t)(b * S_LEN + s) * (NKV * DH) + kvh * DH + d];
}

// ---------------------------------------------------------------------------
// Tiled WMMA GEMM: C[M,N] = act(A[M,K] @ Wt[N,K]^T); A,Wt f16, K mult of 32.
// 128x128 tile, 8 waves, double-buffered LDS fed by async global->LDS copies.
// Wave (wm 0..3, wn 0..1): 32x64 sub-tile, 2 A-frags x 4 B-frags in registers.
// ---------------------------------------------------------------------------
template<typename TC, bool GATE>
__global__ __launch_bounds__(256) void gemm_wmma_kernel(const _Float16* __restrict__ A,
                                                        const _Float16* __restrict__ Wt,
                                                        TC* __restrict__ C,
                                                        int M, int N, int K)
{
    __shared__ __align__(16) _Float16 As[2][128][32];   // A tile row-major (16 KB)
    __shared__ __align__(16) _Float16 Bs[2][128][32];   // Bs[n][k]         (16 KB)

    const int tid  = threadIdx.x;
    const int lane = tid & 31;
    const int wave = tid >> 5;
    const int ln   = lane & 15;
    const int hi   = lane >> 4;
    const int wm   = wave & 3;
    const int wn   = wave >> 2;
    const int bm0  = blockIdx.y * 128;
    const int bn0  = blockIdx.x * 128;

    v8f acc[2][4] = {};

    // 512 16B-chunks per tile; 2 per thread for A and for B.
    auto stage = [&](int kt, int buf) {
        const int k0 = kt * 32;
        #pragma unroll
        for (int j = 0; j < 2; ++j) {
            int c   = tid + j * 256;          // 0..511
            int row = c >> 2, off = (c & 3) * 8;
            copy16_to_lds(A  + (size_t)(bm0 + row) * K + k0 + off, &As[buf][row][off]);
            copy16_to_lds(Wt + (size_t)(bn0 + row) * K + k0 + off, &Bs[buf][row][off]);
        }
    };

    const int nk = K / 32;
    stage(0, 0);

    for (int kt = 0; kt < nk; ++kt) {
        const int buf = kt & 1;
        async_wait_all();            // this wave's tile-kt copies complete
        __syncthreads();             // everyone's copies + prev compute complete
        if (kt + 1 < nk) stage(kt + 1, buf ^ 1);   // overlap with WMMAs below

        FragH af[2], bf[4];
        #pragma unroll
        for (int mt = 0; mt < 2; ++mt) {
            // A-frag: lane<16 K{0..15}; lane>=16 K{16..31}
            const _Float16* ap = &As[buf][wm*32 + mt*16 + ln][hi * 16];
            af[mt].f4[0] = *(const float4*)ap;
            af[mt].f4[1] = *(const float4*)(ap + 8);
        }
        #pragma unroll
        for (int nt = 0; nt < 4; ++nt) {
            // B-frag: lane = column n, K half = hi*16..hi*16+15
            const _Float16* bp = &Bs[buf][wn*64 + nt*16 + ln][hi * 16];
            bf[nt].f4[0] = *(const float4*)bp;
            bf[nt].f4[1] = *(const float4*)(bp + 8);
        }
        #pragma unroll
        for (int mt = 0; mt < 2; ++mt)
            #pragma unroll
            for (int nt = 0; nt < 4; ++nt)
                acc[mt][nt] = WMMA_F32_F16(af[mt].v, bf[nt].v, acc[mt][nt]);
    }

    // C/D layout: VGPR r -> row r (lanes 0-15) / row r+8 (lanes 16-31)
    #pragma unroll
    for (int mt = 0; mt < 2; ++mt)
        #pragma unroll
        for (int nt = 0; nt < 4; ++nt)
            #pragma unroll
            for (int r = 0; r < 8; ++r) {
                int row = bm0 + wm*32 + mt*16 + r + hi*8;
                int col = bn0 + wn*64 + nt*16 + ln;
                float val = acc[mt][nt][r];
                if constexpr (GATE) val = 1.0f / (1.0f + __expf(-val));
                C[(size_t)row * N + col] = (TC)val;
            }
}

// ---------------------------------------------------------------------------
// RoPE on first 64 dims of each head (rotate-half), in-place on f16 buffer.
// ---------------------------------------------------------------------------
__global__ void rope_kernel(_Float16* __restrict__ x, int nheads)
{
    int idx = blockIdx.x * blockDim.x + threadIdx.x;
    int total = MROWS * nheads * 32;
    if (idx >= total) return;
    int i    = idx & 31;
    int head = (idx >> 5) % nheads;
    int row  = idx / (32 * nheads);
    int pos  = row % S_LEN;
    float inv_freq = __expf(-(float)(2 * i) * (9.210340371976184f / 64.0f));
    float ang = (float)pos * inv_freq;
    float s, c;
    __sincosf(ang, &s, &c);
    _Float16* p = x + (size_t)row * (nheads * DH) + head * DH;
    float x1 = (float)p[i];
    float x2 = (float)p[i + 32];
    p[i]      = (_Float16)(x1 * c - x2 * s);
    p[i + 32] = (_Float16)(x2 * c + x1 * s);
}

// ---------------------------------------------------------------------------
// Flash attention + fused gate. 8 waves x 16 query rows. Q tile staged once
// in LDS (async); K (row-major) and V (pre-transposed d-major) 32-key blocks
// double-buffered via async global->LDS. Scores: 16 WMMAs (D=256); online
// softmax with 16-lane butterflies; P re-shaped through a wave-private LDS
// tile; P@V: 16 WMMAs. Q-in-LDS keeps the wave under 256 VGPRs.
// ---------------------------------------------------------------------------
__global__ __launch_bounds__(256) void attn_wmma_kernel(const _Float16* __restrict__ q,
                                                        const _Float16* __restrict__ k,
                                                        const _Float16* __restrict__ vt,
                                                        const _Float16* __restrict__ gate,
                                                        _Float16* __restrict__ out)
{
    __shared__ __align__(16) _Float16 Qs[128][DH];      // 64 KB (whole block's Q)
    __shared__ __align__(16) _Float16 Ks[2][32][DH];    // 32 KB
    __shared__ __align__(16) _Float16 Vt[2][DH][32];    // 32 KB
    __shared__ __align__(16) _Float16 Ps[8][16][32];    //  8 KB

    const int b    = blockIdx.z;
    const int h    = blockIdx.y;
    const int kvh  = h / GQ;
    const int tid  = threadIdx.x;
    const int lane = tid & 31;
    const int wave = tid >> 5;
    const int ln   = lane & 15;
    const int hi   = lane >> 4;
    const int qb0  = blockIdx.x * 128;
    const int q0   = qb0 + wave * 16;
    const int QSTR = NH * DH;    // 4096
    const int KSTR = NKV * DH;   // 512

    // ---- stage whole Q tile (128 x 256 f16 = 4096 chunks, 16/thread) ----
    {
        const _Float16* qg = q + (size_t)(b * S_LEN + qb0) * QSTR + h * DH;
        #pragma unroll
        for (int j = 0; j < 16; ++j) {
            int c   = (tid << 4) | j;              // 0..4095
            int row = c >> 5, off = (c & 31) * 8;
            copy16_to_lds(qg + (size_t)row * QSTR + off, &Qs[row][off]);
        }
    }

    v8f acc[16] = {};
    float mst[8], lst[8];
    #pragma unroll
    for (int r = 0; r < 8; ++r) { mst[r] = -1e30f; lst[r] = 0.0f; }

    const int nkv    = blockIdx.x * 4 + 4;   // block-uniform causal kv-block count
    const int my_nkv = q0 / 32 + 1;          // this wave's causal limit
    const _Float16* kbase  = k  + (size_t)(b * S_LEN) * KSTR + kvh * DH;
    const _Float16* vtbase = vt + (size_t)(b * NKV + kvh) * DH * S_LEN;

    // 1024 16B-chunks each for K and V per 32-key block: 4+4 per thread.
    auto stageKV = [&](int kb, int buf) {
        const int kb32 = kb * 32;
        #pragma unroll
        for (int j = 0; j < 4; ++j) {
            int c = (tid << 2) | j;                 // 0..1023
            int krow = c >> 5, koff = (c & 31) * 8;
            copy16_to_lds(kbase + (size_t)(kb32 + krow) * KSTR + koff,
                          &Ks[buf][krow][koff]);
        }
        #pragma unroll
        for (int j = 0; j < 4; ++j) {
            int c = (tid << 2) | j;
            int d = c >> 2, soff = (c & 3) * 8;
            copy16_to_lds(vtbase + (size_t)d * S_LEN + kb32 + soff,
                          &Vt[buf][d][soff]);
        }
    };

    stageKV(0, 0);

    for (int kb = 0; kb < nkv; ++kb) {
        const int buf = kb & 1;
        async_wait_all();            // Q (first iter) + tile-kb K/V complete
        __syncthreads();
        if (kb + 1 < nkv) stageKV(kb + 1, buf ^ 1);   // overlap with compute

        if (kb < my_nkv) {
            // ---- scores: S = Q @ K^T over D=256 (8 k-chunks x 2 key tiles) ----
            v8f sc[2] = {};
            #pragma unroll
            for (int c = 0; c < 8; ++c) {
                FragH af;
                const _Float16* ap = &Qs[wave * 16 + ln][c * 32 + hi * 16];
                af.f4[0] = *(const float4*)ap;
                af.f4[1] = *(const float4*)(ap + 8);
                #pragma unroll
                for (int t = 0; t < 2; ++t) {
                    FragH bfv;
                    const _Float16* kp = &Ks[buf][t * 16 + ln][c * 32 + hi * 16];
                    bfv.f4[0] = *(const float4*)kp;
                    bfv.f4[1] = *(const float4*)(kp + 8);
                    sc[t] = WMMA_F32_F16(af.v, bfv.v, sc[t]);
                }
            }

            // ---- online softmax ----
            #pragma unroll
            for (int r = 0; r < 8; ++r) {
                const int qg = q0 + r + hi * 8;
                float s0 = sc[0][r] * 0.0625f;            // D^-0.5
                float s1 = sc[1][r] * 0.0625f;
                if (kb * 32 + ln      > qg) s0 = -1e30f;  // causal mask
                if (kb * 32 + 16 + ln > qg) s1 = -1e30f;
                float rm = fmaxf(s0, s1);
                #pragma unroll
                for (int off = 1; off < 16; off <<= 1)
                    rm = fmaxf(rm, __shfl_xor(rm, off, 32));
                float mnew  = fmaxf(mst[r], rm);
                float alpha = __expf(mst[r] - mnew);
                float p0 = __expf(s0 - mnew);
                float p1 = __expf(s1 - mnew);
                float rs = p0 + p1;
                #pragma unroll
                for (int off = 1; off < 16; off <<= 1)
                    rs += __shfl_xor(rs, off, 32);
                lst[r] = lst[r] * alpha + rs;
                mst[r] = mnew;
                #pragma unroll
                for (int nt = 0; nt < 16; ++nt) acc[nt][r] *= alpha;
                Ps[wave][r + hi * 8][ln]      = (_Float16)p0;
                Ps[wave][r + hi * 8][16 + ln] = (_Float16)p1;
            }

            // ---- P @ V (wave-private LDS round trip for layout change) ----
            FragH pf;
            const _Float16* pp = &Ps[wave][ln][hi * 16];
            pf.f4[0] = *(const float4*)pp;
            pf.f4[1] = *(const float4*)(pp + 8);
            #pragma unroll
            for (int nt = 0; nt < 16; ++nt) {
                FragH vf;
                const _Float16* vp = &Vt[buf][nt * 16 + ln][hi * 16];
                vf.f4[0] = *(const float4*)vp;
                vf.f4[1] = *(const float4*)(vp + 8);
                acc[nt] = WMMA_F32_F16(pf.v, vf.v, acc[nt]);
            }
        }
    }

    // ---- epilogue: normalize, gate, store f16 ----
    const _Float16* grow = gate + (size_t)(b * S_LEN + q0) * QSTR + h * DH;
    _Float16*       orow = out  + (size_t)(b * S_LEN + q0) * QSTR + h * DH;
    #pragma unroll
    for (int r = 0; r < 8; ++r) {
        const float invl = 1.0f / lst[r];
        const int   row  = r + hi * 8;
        #pragma unroll
        for (int nt = 0; nt < 16; ++nt) {
            const int col = nt * 16 + ln;
            float g = (float)grow[(size_t)row * QSTR + col];
            orow[(size_t)row * QSTR + col] = (_Float16)(acc[nt][r] * invl * g);
        }
    }
}

// ---------------------------------------------------------------------------
extern "C" void kernel_launch(void* const* d_in, const int* in_sizes, int n_in,
                              void* d_out, int out_size, void* d_ws, size_t ws_size,
                              hipStream_t stream)
{
    const float* hs = (const float*)d_in[0];
    const float* Wq = (const float*)d_in[1];
    const float* Wk = (const float*)d_in[2];
    const float* Wv = (const float*)d_in[3];
    const float* Wg = (const float*)d_in[4];
    const float* Wo = (const float*)d_in[5];
    float* outp = (float*)d_out;

    // workspace layout (all f16)
    _Float16* hs16 = (_Float16*)d_ws;                       // 4096 x 2048
    _Float16* wqt  = hs16 + (size_t)MROWS * E_DIM;          // 4096 x 2048 (N x K)
    _Float16* wkt  = wqt  + (size_t)(NH  * DH) * E_DIM;     //  512 x 2048
    _Float16* wvt  = wkt  + (size_t)(NKV * DH) * E_DIM;     //  512 x 2048
    _Float16* wgt  = wvt  + (size_t)(NKV * DH) * E_DIM;     // 4096 x 2048
    _Float16* wot  = wgt  + (size_t)(NH  * DH) * E_DIM;     // 2048 x 4096 (N x K)
    _Float16* qbuf = wot  + (size_t)E_DIM * (NH * DH);      // 4096 x 4096
    _Float16* kbuf = qbuf + (size_t)MROWS * (NH  * DH);     // 4096 x  512
    _Float16* vbuf = kbuf + (size_t)MROWS * (NKV * DH);     // 4096 x  512
    _Float16* gbuf = vbuf + (size_t)MROWS * (NKV * DH);     // 4096 x 4096
    _Float16* abuf = gbuf + (size_t)MROWS * (NH  * DH);     // 4096 x 4096
    _Float16* vtb  = abuf + (size_t)MROWS * (NH  * DH);     // (2*2*256) x 2048

    dim3 blk(256);

    // prepass: f16 conversion + weight transposes (one-time, bandwidth-trivial)
    {
        long n = (long)MROWS * E_DIM;
        convert_f16_kernel<<<dim3((unsigned)((n + 255) / 256)), blk, 0, stream>>>(hs, hs16, n);
    }
    auto tlaunch = [&](const float* W, _Float16* Wt, int K, int N) {
        long n = (long)K * N;
        transpose_f16_kernel<<<dim3((unsigned)((n + 255) / 256)), blk, 0, stream>>>(W, Wt, K, N);
    };
    tlaunch(Wq, wqt, E_DIM, NH * DH);
    tlaunch(Wk, wkt, E_DIM, NKV * DH);
    tlaunch(Wv, wvt, E_DIM, NKV * DH);
    tlaunch(Wg, wgt, E_DIM, NH * DH);
    tlaunch(Wo, wot, NH * DH, E_DIM);

    // projections (f16 WMMA, f32 accumulate)
    gemm_wmma_kernel<_Float16, false>
        <<<dim3((NH * DH) / 128, MROWS / 128), blk, 0, stream>>>(hs16, wqt, qbuf, MROWS, NH * DH, E_DIM);
    gemm_wmma_kernel<_Float16, false>
        <<<dim3((NKV * DH) / 128, MROWS / 128), blk, 0, stream>>>(hs16, wkt, kbuf, MROWS, NKV * DH, E_DIM);
    gemm_wmma_kernel<_Float16, false>
        <<<dim3((NKV * DH) / 128, MROWS / 128), blk, 0, stream>>>(hs16, wvt, vbuf, MROWS, NKV * DH, E_DIM);
    gemm_wmma_kernel<_Float16, true>
        <<<dim3((NH * DH) / 128, MROWS / 128), blk, 0, stream>>>(hs16, wgt, gbuf, MROWS, NH * DH, E_DIM);

    rope_kernel<<<(MROWS * NH  * 32 + 255) / 256, blk, 0, stream>>>(qbuf, NH);
    rope_kernel<<<(MROWS * NKV * 32 + 255) / 256, blk, 0, stream>>>(kbuf, NKV);

    {
        long n = (long)NB * NKV * DH * S_LEN;
        vtranspose_kernel<<<dim3((unsigned)((n + 255) / 256)), blk, 0, stream>>>(vbuf, vtb);
    }

    attn_wmma_kernel<<<dim3(S_LEN / 128, NH, NB), blk, 0, stream>>>(qbuf, kbuf, vtb, gbuf, abuf);

    gemm_wmma_kernel<float, false>
        <<<dim3(E_DIM / 128, MROWS / 128), blk, 0, stream>>>(abuf, wot, outp, MROWS, E_DIM, NH * DH);
}
